// InteractionModule_65377992180232
// MI455X (gfx1250) — matrix operand order repaired and checked by reading, not compiled, not run on hardware
//
#include <hip/hip_runtime.h>
#include <hip/hip_bf16.h>
#include <math.h>

// ---------------------------------------------------------------------------
// SpookyNet interaction block on MI455X (gfx1250, wave32, WMMA bf16->f32).
// All GEMMs use v_wmma_f32_16x16x32_bf16. Pair phase fuses the radial WMMA
// GEMM with the 9-way atomic scatter. Attention exploits sorted batch_seg:
// per-segment WMMA GEMMs for KV = Kp^T V and out = Qp KV.
// ---------------------------------------------------------------------------

typedef __bf16 bf16_t;
typedef __attribute__((ext_vector_type(16))) __bf16 v16bf;
typedef __attribute__((ext_vector_type(2)))  __bf16 v2bf;
typedef __attribute__((ext_vector_type(8)))  float  v8f;

#define WMMA_BF16(A, B, C) \
  __builtin_amdgcn_wmma_f32_16x16x32_bf16(false, (A), false, (B), (short)0, (C), false, false)

#if defined(__has_builtin)
#if __has_builtin(__builtin_amdgcn_cvt_pk_bf16_f32)
#define HAVE_CVT_PK_BF16 1
#endif
#endif
#ifndef HAVE_CVT_PK_BF16
#define HAVE_CVT_PK_BF16 0
#endif

static constexpr int   NN  = 8192;
static constexpr int   PP_ = 131072;
static constexpr int   BB  = 8;
static constexpr size_t NF = (size_t)NN * 128;

// ------------------------------- helpers -----------------------------------

__device__ __forceinline__ bf16_t f2bf(float f) {
  union { float f; unsigned u; } v; v.f = f;
  unsigned r = v.u + 0x7FFFu + ((v.u >> 16) & 1u);        // RNE
  unsigned short h = (unsigned short)(r >> 16);
  return __builtin_bit_cast(bf16_t, h);
}
__device__ __forceinline__ bf16_t bfz() {
  return __builtin_bit_cast(bf16_t, (unsigned short)0);
}
// pack two floats into frag elements i, i+1 (v_cvt_pk_bf16_f32 when available)
__device__ __forceinline__ void put2(v16bf& f, int i, float x, float y) {
#if HAVE_CVT_PK_BF16
  v2bf p = __builtin_amdgcn_cvt_pk_bf16_f32(x, y);
  f[i] = p[0]; f[i + 1] = p[1];
#else
  f[i] = f2bf(x); f[i + 1] = f2bf(y);
#endif
}
__device__ __forceinline__ void swish4(float4& x, const float4 a, const float4 b) {
  x.x = a.x * x.x / (1.0f + __expf(-b.x * x.x));
  x.y = a.y * x.y / (1.0f + __expf(-b.y * x.y));
  x.z = a.z * x.z / (1.0f + __expf(-b.z * x.z));
  x.w = a.w * x.w / (1.0f + __expf(-b.w * x.w));
}

// Build one 16x16x32 A fragment (bf16) from an f32 row-major matrix with
// ld = 128, optionally applying learnable swish (per-K a,b vectors).
// A layout: lane&15 = M row; lo half K = {kb..kb+15},
// hi half K = {kb+16..kb+31}.
template <bool SWISH>
__device__ __forceinline__ v16bf load_a_frag(const float* __restrict__ X, int mrow, int kb,
                                             const float* __restrict__ sa,
                                             const float* __restrict__ sb, bool lo) {
  int c0 = kb + (lo ? 0 : 16);
  const float* xr = X + mrow * 128;
  float4 x0 = *(const float4*)(xr + c0);
  float4 x1 = *(const float4*)(xr + c0 + 4);
  float4 x2 = *(const float4*)(xr + c0 + 8);
  float4 x3 = *(const float4*)(xr + c0 + 12);
  if (SWISH) {
    float4 a0 = *(const float4*)(sa + c0), a1 = *(const float4*)(sa + c0 + 4);
    float4 a2 = *(const float4*)(sa + c0 + 8), a3 = *(const float4*)(sa + c0 + 12);
    float4 b0 = *(const float4*)(sb + c0), b1 = *(const float4*)(sb + c0 + 4);
    float4 b2 = *(const float4*)(sb + c0 + 8), b3 = *(const float4*)(sb + c0 + 12);
    swish4(x0, a0, b0); swish4(x1, a1, b1); swish4(x2, a2, b2); swish4(x3, a3, b3);
  }
  v16bf f;
  put2(f, 0,  x0.x, x0.y); put2(f, 2,  x0.z, x0.w);
  put2(f, 4,  x1.x, x1.y); put2(f, 6,  x1.z, x1.w);
  put2(f, 8,  x2.x, x2.y); put2(f, 10, x2.z, x2.w);
  put2(f, 12, x3.x, x3.y); put2(f, 14, x3.z, x3.w);
  return f;
}
__device__ __forceinline__ v16bf zero_frag() {
  v16bf z;
#pragma unroll
  for (int t = 0; t < 16; ++t) z[t] = bfz();
  return z;
}

// ------------------------- generic K=128 GEMM ------------------------------
// Y[m, col] = sum_k act(X[m,k]) * Wb[col][k] (+bias[col]) (+res[m,col])
// Wb is bf16 [Ncols][128] row-major (i.e. B = W^T: contiguous K per column).
// grid = (M/64, Ncols/128), block = 128 (4 waves; wave -> 16 rows x 128 cols).
template <bool SWISH, bool BIAS, bool RES, int LDO>
__global__ void __launch_bounds__(128) gemm_kernel(
    const float* __restrict__ X, const bf16_t* __restrict__ Wb,
    const float* __restrict__ bias, const float* __restrict__ sa,
    const float* __restrict__ sb, const float* __restrict__ res,
    float* __restrict__ Y) {
  int w = threadIdx.x >> 5, lane = threadIdx.x & 31;
  bool lo = lane < 16;
  int m0 = (blockIdx.x * 4 + w) * 16;
  int mrow = m0 + (lane & 15);
  v16bf a0 = load_a_frag<SWISH>(X, mrow,  0, sa, sb, lo);
  v16bf a1 = load_a_frag<SWISH>(X, mrow, 32, sa, sb, lo);
  v16bf a2 = load_a_frag<SWISH>(X, mrow, 64, sa, sb, lo);
  v16bf a3 = load_a_frag<SWISH>(X, mrow, 96, sa, sb, lo);
  int colbase = blockIdx.y * 128;
#pragma unroll
  for (int nt = 0; nt < 8; ++nt) {
    int col = colbase + nt * 16 + (lane & 15);
    v8f acc;
    float cinit = BIAS ? bias[col] : 0.0f;
#pragma unroll
    for (int r = 0; r < 8; ++r) acc[r] = cinit;
    const bf16_t* wr = Wb + col * 128 + (lo ? 0 : 16);
    acc = WMMA_BF16(a0, *(const v16bf*)(wr +  0), acc);
    acc = WMMA_BF16(a1, *(const v16bf*)(wr + 32), acc);
    acc = WMMA_BF16(a2, *(const v16bf*)(wr + 64), acc);
    acc = WMMA_BF16(a3, *(const v16bf*)(wr + 96), acc);
    int rbase = m0 + (lo ? 0 : 8);
    float* yr = Y + rbase * LDO + col;
    const float* rr = RES ? (res + rbase * LDO + col) : nullptr;
#pragma unroll
    for (int r = 0; r < 8; ++r) {
      float v = acc[r];
      if (RES) v += rr[r * LDO];
      yr[r * LDO] = v;
    }
  }
}

// -------------------- fused radial GEMM + pair scatter ---------------------
// Per wave: 16 pairs. A = rbf rows (K=16 zero-padded to 32, bf16); for each of
// the 3 radial weights and 8 feature tiles, one WMMA yields g[16x16], which is
// scattered with atomics: s += gs*xs[j], pp[k] += gp*pij[k]*xp[j],
// dd[k] += gd*dij[k]*xd[j].
__global__ void __launch_bounds__(128) pair_kernel(
    const float* __restrict__ rbf, const int* __restrict__ idx_i,
    const int* __restrict__ idx_j, const float* __restrict__ pij,
    const float* __restrict__ dij, const float* __restrict__ xs,
    const float* __restrict__ xp, const float* __restrict__ xd,
    const bf16_t* __restrict__ rsb, const bf16_t* __restrict__ rpb,
    const bf16_t* __restrict__ rdb, float* __restrict__ s_acc,
    float* __restrict__ pp, float* __restrict__ dd) {
  int w = threadIdx.x >> 5, lane = threadIdx.x & 31;
  bool lo = lane < 16;
  int e0 = (blockIdx.x * 4 + w) * 16;
  int erow = e0 + (lane & 15);
  // A fragment: lo -> K{0..15} real; hi -> K{16..31} pad.
  v16bf a;
  if (lo) {
    const float* rr = rbf + erow * 16;
    float4 r0 = *(const float4*)(rr);
    float4 r1 = *(const float4*)(rr + 4);
    float4 r2 = *(const float4*)(rr + 8);
    float4 r3 = *(const float4*)(rr + 12);
    put2(a, 0, r0.x, r0.y); put2(a, 2, r0.z, r0.w);
    put2(a, 4, r1.x, r1.y); put2(a, 6, r1.z, r1.w);
    put2(a, 8, r2.x, r2.y); put2(a, 10, r2.z, r2.w);
    put2(a, 12, r3.x, r3.y); put2(a, 14, r3.z, r3.w);
  } else {
#pragma unroll
    for (int t = 0; t < 16; ++t) a[t] = bfz();
  }
  int ii[8], jj[8];
  float pj[8][3], dj[8][5];
#pragma unroll
  for (int r = 0; r < 8; ++r) {
    int e = e0 + r + (lo ? 0 : 8);
    ii[r] = idx_i[e]; jj[r] = idx_j[e];
#pragma unroll
    for (int k = 0; k < 3; ++k) pj[r][k] = pij[e * 3 + k];
#pragma unroll
    for (int k = 0; k < 5; ++k) dj[r][k] = dij[e * 5 + k];
  }
#pragma unroll 1
  for (int nt = 0; nt < 8; ++nt) {
    int f = nt * 16 + (lane & 15);
    { // gs -> s
      v8f acc; for (int r = 0; r < 8; ++r) acc[r] = 0.0f;
      acc = WMMA_BF16(a, *(const v16bf*)(rsb + f * 32 + (lo ? 0 : 16)), acc);
#pragma unroll
      for (int r = 0; r < 8; ++r) {
        float v = acc[r] * xs[jj[r] * 128 + f];
        unsafeAtomicAdd(&s_acc[ii[r] * 128 + f], v);
      }
    }
    { // gp -> pp[3]
      v8f acc; for (int r = 0; r < 8; ++r) acc[r] = 0.0f;
      acc = WMMA_BF16(a, *(const v16bf*)(rpb + f * 32 + (lo ? 0 : 16)), acc);
#pragma unroll
      for (int r = 0; r < 8; ++r) {
        float gx = acc[r] * xp[jj[r] * 128 + f];
#pragma unroll
        for (int k = 0; k < 3; ++k)
          unsafeAtomicAdd(&pp[(ii[r] * 3 + k) * 128 + f], gx * pj[r][k]);
      }
    }
    { // gd -> dd[5]
      v8f acc; for (int r = 0; r < 8; ++r) acc[r] = 0.0f;
      acc = WMMA_BF16(a, *(const v16bf*)(rdb + f * 32 + (lo ? 0 : 16)), acc);
#pragma unroll
      for (int r = 0; r < 8; ++r) {
        float gx = acc[r] * xd[jj[r] * 128 + f];
#pragma unroll
        for (int k = 0; k < 5; ++k)
          unsafeAtomicAdd(&dd[(ii[r] * 5 + k) * 128 + f], gx * dj[r][k]);
      }
    }
  }
}

// --------------------------- attention kernels -----------------------------

__global__ void seg_offsets_kernel(const int* __restrict__ bs, int* __restrict__ offs) {
  if (blockIdx.x == 0 && threadIdx.x == 0) {
    offs[0] = 0;
    int p = 0;
    for (int b = 1; b < BB; ++b) {
      while (p < NN && bs[p] < b) ++p;
      offs[b] = p;
    }
    offs[BB] = NN;
  }
}

// per-row stats: h = sum(x^2)/(2 sqrt(128)), rowmax of U (for q and k)
__global__ void __launch_bounds__(128) rowstats_kernel(
    const float* __restrict__ q, const float* __restrict__ k,
    const float* __restrict__ Uq, const float* __restrict__ Uk,
    float* __restrict__ hq, float* __restrict__ hk,
    float* __restrict__ rmq, float* __restrict__ rmk) {
  int n = blockIdx.x * 4 + (threadIdx.x >> 5);
  int lane = threadIdx.x & 31;
  float sq = 0.f, sk = 0.f, mq = -3.402823466e38f, mk = -3.402823466e38f;
  for (int t = lane; t < 128; t += 32) {
    float qv = q[n * 128 + t];  sq += qv * qv;
    float kv = k[n * 128 + t];  sk += kv * kv;
    mq = fmaxf(mq, Uq[n * 128 + t]);
    mk = fmaxf(mk, Uk[n * 128 + t]);
  }
#pragma unroll
  for (int off = 16; off; off >>= 1) {
    sq += __shfl_xor(sq, off); sk += __shfl_xor(sk, off);
    mq = fmaxf(mq, __shfl_xor(mq, off)); mk = fmaxf(mk, __shfl_xor(mk, off));
  }
  if (lane == 0) {
    float inv = 1.0f / (2.0f * sqrtf(128.0f));
    hq[n] = sq * inv; hk[n] = sk * inv; rmq[n] = mq; rmk[n] = mk;
  }
}

__global__ void __launch_bounds__(256) segmax_kernel(
    const float* __restrict__ rmk, const int* __restrict__ offs, float* __restrict__ smax) {
  int b = threadIdx.x >> 5, lane = threadIdx.x & 31;
  int s0 = offs[b], s1 = offs[b + 1];
  float m = -3.402823466e38f;
  for (int n = s0 + lane; n < s1; n += 32) m = fmaxf(m, rmk[n]);
#pragma unroll
  for (int off = 16; off; off >>= 1) m = fmaxf(m, __shfl_xor(m, off));
  if (lane == 0) smax[b] = m;
}

__global__ void __launch_bounds__(256) phiq_kernel(
    const float* __restrict__ Uq, const float* __restrict__ hq,
    const float* __restrict__ rmq, float* __restrict__ Qp) {
  int idx = blockIdx.x * 256 + threadIdx.x;
  int n = idx >> 7;
  Qp[idx] = (__expf(Uq[idx] - hq[n] - rmq[n]) + 1e-4f) * 0.08838834764831845f; // 1/sqrt(128)
}
__global__ void __launch_bounds__(256) phik_kernel(
    const float* __restrict__ Uk, const float* __restrict__ hk,
    const float* __restrict__ smax, const int* __restrict__ bseg,
    float* __restrict__ Kp, bf16_t* __restrict__ KpT) {
  int idx = blockIdx.x * 256 + threadIdx.x;
  int n = idx >> 7, m = idx & 127;
  float v = (__expf(Uk[idx] - hk[n] - smax[bseg[n]]) + 1e-4f) * 0.08838834764831845f;
  Kp[idx] = v;
  KpT[m * NN + n] = f2bf(v);
}
__global__ void __launch_bounds__(256) vt_kernel(const float* __restrict__ V,
                                                 bf16_t* __restrict__ VT) {
  int idx = blockIdx.x * 256 + threadIdx.x;
  int n = idx >> 7, f = idx & 127;
  VT[f * NN + n] = f2bf(V[idx]);
}
__global__ void __launch_bounds__(128) ksum_kernel(const float* __restrict__ Kp,
                                                   const int* __restrict__ offs,
                                                   float* __restrict__ Ksum) {
  int b = blockIdx.x, m = threadIdx.x;
  float s = 0.f;
  for (int n = offs[b]; n < offs[b + 1]; ++n) s += Kp[n * 128 + m];
  Ksum[b * 128 + m] = s;
}
__global__ void __launch_bounds__(256) norm_kernel(const float* __restrict__ Qp,
                                                   const float* __restrict__ Ksum,
                                                   const int* __restrict__ bseg,
                                                   float* __restrict__ nv) {
  int n = blockIdx.x * 256 + threadIdx.x;
  int b = bseg[n];
  float s = 0.f;
  for (int m = 0; m < 128; ++m) s += Qp[n * 128 + m] * Ksum[b * 128 + m];
  nv[n] = s + 1e-8f;
}

// KV[b][m][f] = sum_{n in seg b} Kp[n][m] * V[n][f], stored transposed as
// KVT[b][f][m] (bf16) so the next GEMM's B-loads are contiguous.
// grid = B blocks, 8 waves each (wave w -> m tile w, all 8 f tiles).
// A fragment hoisted per K-step; un-guarded fast path inside the segment.
__global__ void __launch_bounds__(256) kv_kernel(
    const bf16_t* __restrict__ KpT, const bf16_t* __restrict__ VT,
    const int* __restrict__ offs, bf16_t* __restrict__ KVT) {
  int b = blockIdx.x, w = threadIdx.x >> 5, lane = threadIdx.x & 31;
  bool lo = lane < 16;
  int s0 = offs[b], s1 = offs[b + 1];
  int m0 = w * 16, mrow = m0 + (lane & 15);
  v8f acc[8];
#pragma unroll
  for (int nt = 0; nt < 8; ++nt)
    for (int r = 0; r < 8; ++r) acc[nt][r] = 0.0f;
  const bf16_t* arow = KpT + mrow * NN;
  for (int kb = s0; kb < s1; kb += 32) {
    bool full = (kb + 32 <= s1);
    v16bf av;
    int ka0 = kb + (lo ? 0 : 16);
    if (full) {
#pragma unroll
      for (int t = 0; t < 8; ++t) { av[t] = arow[ka0 + t]; av[t + 8] = arow[ka0 + 8 + t]; }
    } else {
#pragma unroll
      for (int t = 0; t < 8; ++t) {
        int k1 = ka0 + t, k2 = ka0 + 8 + t;
        av[t]     = (k1 < s1) ? arow[k1] : bfz();
        av[t + 8] = (k2 < s1) ? arow[k2] : bfz();
      }
    }
    int kb0 = kb + (lo ? 0 : 16);
#pragma unroll
    for (int nt = 0; nt < 8; ++nt) {
      const bf16_t* brow = VT + (nt * 16 + (lane & 15)) * NN;
      v16bf bv;
      if (full) {
#pragma unroll
        for (int t = 0; t < 16; ++t) bv[t] = brow[kb0 + t];
      } else {
#pragma unroll
        for (int t = 0; t < 16; ++t) { int k = kb0 + t; bv[t] = (k < s1) ? brow[k] : bfz(); }
      }
      acc[nt] = WMMA_BF16(av, bv, acc[nt]);
    }
  }
#pragma unroll
  for (int nt = 0; nt < 8; ++nt) {
    int f = nt * 16 + (lane & 15);
#pragma unroll
    for (int r = 0; r < 8; ++r) {
      int m = m0 + r + (lo ? 0 : 8);
      KVT[(b * 128 + f) * 128 + m] = f2bf(acc[nt][r]);
    }
  }
}

// nl[n][f] = (Qp[n,:] . KV[b][:,f]) / norm[n], per segment.
// grid = (B, 128), block 128 (4 waves, each a 16-row tile inside segment b).
__global__ void __launch_bounds__(128) att_kernel(
    const float* __restrict__ Qp, const bf16_t* __restrict__ KVT,
    const float* __restrict__ nv, const int* __restrict__ offs,
    float* __restrict__ nl) {
  int b = blockIdx.x;
  int tile = blockIdx.y * 4 + (threadIdx.x >> 5);
  int lane = threadIdx.x & 31;
  bool lo = lane < 16;
  int s0 = offs[b], s1 = offs[b + 1];
  int m0 = s0 + tile * 16;
  if (m0 >= s1) return;
  int mrow = m0 + (lane & 15);
  v16bf a0, a1, a2, a3;
  if (mrow < s1) {
    a0 = load_a_frag<false>(Qp, mrow,  0, nullptr, nullptr, lo);
    a1 = load_a_frag<false>(Qp, mrow, 32, nullptr, nullptr, lo);
    a2 = load_a_frag<false>(Qp, mrow, 64, nullptr, nullptr, lo);
    a3 = load_a_frag<false>(Qp, mrow, 96, nullptr, nullptr, lo);
  } else {
    a0 = a1 = a2 = a3 = zero_frag();
  }
  const bf16_t* KVb = KVT + b * 128 * 128;
#pragma unroll 1
  for (int nt = 0; nt < 8; ++nt) {
    int f = nt * 16 + (lane & 15);
    const bf16_t* wr = KVb + f * 128 + (lo ? 0 : 16);
    v8f acc; for (int r = 0; r < 8; ++r) acc[r] = 0.0f;
    acc = WMMA_BF16(a0, *(const v16bf*)(wr +  0), acc);
    acc = WMMA_BF16(a1, *(const v16bf*)(wr + 32), acc);
    acc = WMMA_BF16(a2, *(const v16bf*)(wr + 64), acc);
    acc = WMMA_BF16(a3, *(const v16bf*)(wr + 96), acc);
#pragma unroll
    for (int r = 0; r < 8; ++r) {
      int m = m0 + r + (lo ? 0 : 8);
      if (m < s1) nl[m * 128 + f] = acc[r] / nv[m];
    }
  }
}

// ------------------------- small utility kernels ---------------------------

__global__ void __launch_bounds__(256) cvt_kernel(const float* __restrict__ s,
                                                  bf16_t* __restrict__ d, int n) {
  int i = blockIdx.x * 256 + threadIdx.x;
  if (i < n) d[i] = f2bf(s[i]);
}
// omega is used un-transposed (X @ omega), so store transposed + folded scale.
__global__ void __launch_bounds__(256) cvt_omega_kernel(const float* __restrict__ om,
                                                        bf16_t* __restrict__ d, float sc) {
  int i = blockIdx.x * 256 + threadIdx.x;   // i = m*128 + k
  int m = i >> 7, k = i & 127;
  d[i] = f2bf(om[k * 128 + m] * sc);
}
// radial [128][16] -> bf16 [128][32] zero-padded (K padded for 16x16x32 WMMA)
__global__ void __launch_bounds__(256) cvt_radial_kernel(const float* __restrict__ s,
                                                         bf16_t* __restrict__ d) {
  int i = blockIdx.x * 256 + threadIdx.x;   // 128*32
  int f = i >> 5, k = i & 31;
  d[i] = f2bf(k < 16 ? s[f * 16 + k] : 0.0f);
}
__global__ void __launch_bounds__(256) zero_kernel(float* p, int n) {
  int i = blockIdx.x * 256 + threadIdx.x;
  if (i < n) p[i] = 0.0f;
}
__global__ void __launch_bounds__(256) copy_kernel(const float* __restrict__ s,
                                                   float* __restrict__ d, int n) {
  int i = blockIdx.x * 256 + threadIdx.x;
  if (i < n) d[i] = s[i];
}
__global__ void __launch_bounds__(256) add3_kernel(const float* __restrict__ a,
                                                   const float* __restrict__ b,
                                                   const float* __restrict__ c,
                                                   float* __restrict__ y, int n) {
  int i = blockIdx.x * 256 + threadIdx.x;
  if (i < n) y[i] = a[i] + b[i] + c[i];
}
// u = s + sum_k pa*pb + sum_k da*db  (pa/pb packed in pt[.,256], da/db in dt)
__global__ void __launch_bounds__(256) combine_u_kernel(
    const float* __restrict__ s_acc, const float* __restrict__ pt,
    const float* __restrict__ dt, float* __restrict__ u) {
  int idx = blockIdx.x * 256 + threadIdx.x;
  int n = idx >> 7, f = idx & 127;
  float v = s_acc[idx];
#pragma unroll
  for (int k = 0; k < 3; ++k) {
    const float* row = pt + (n * 3 + k) * 256;
    v += row[f] * row[128 + f];
  }
#pragma unroll
  for (int k = 0; k < 5; ++k) {
    const float* row = dt + (n * 5 + k) * 256;
    v += row[f] * row[128 + f];
  }
  u[idx] = v;
}

// ------------------------------- host side ---------------------------------

struct MlpP { const float *W,*a,*b,*W1,*W2,*a1,*a2,*b1,*b2,*c1,*c2,*c; };
struct BlkP { const float *W1,*W2,*a1,*a2,*b1,*b2,*c1,*c2; };
struct MlpB { bf16_t *W1b,*W2b,*Wb; };

extern "C" void kernel_launch(void* const* d_in, const int* in_sizes, int n_in,
                              void* d_out, int out_size, void* d_ws, size_t ws_size,
                              hipStream_t stream) {
  (void)in_sizes; (void)n_in; (void)out_size; (void)ws_size;
  auto F32 = [&](int i) { return (const float*)d_in[i]; };
  auto I32 = [&](int i) { return (const int*)d_in[i]; };

  // top-level inputs (setup_inputs insertion order)
  const float* x_in  = F32(0);
  const float* rbf   = F32(1);
  const float* pij   = F32(2);
  const float* dij   = F32(3);
  const int*   idx_i = I32(4);
  const int*   idx_j = I32(5);
  const int*   bseg  = I32(7);

  // params: jax tree flatten order (dict keys sorted recursively)
  const float* omega = F32(8);
  const float* projd = F32(9);
  const float* projp = F32(10);
  const float* radd  = F32(11);
  const float* radp  = F32(12);
  const float* rads  = F32(13);
  auto mlp_at = [&](int b) -> MlpP {
    return { F32(b+0), F32(b+1), F32(b+2), F32(b+3), F32(b+4), F32(b+5),
             F32(b+6), F32(b+7), F32(b+8), F32(b+9), F32(b+10), F32(b+11) };
  };
  auto blk_at = [&](int b) -> BlkP {
    return { F32(b+0), F32(b+1), F32(b+2), F32(b+3), F32(b+4), F32(b+5),
             F32(b+6), F32(b+7) };
  };
  MlpP m_d = mlp_at(14),  m_k = mlp_at(26),  m_loc = mlp_at(38);
  MlpP m_out = mlp_at(50), m_p = mlp_at(62);
  BlkP b_post = blk_at(74), b_pre = blk_at(82);
  MlpP m_q = mlp_at(90),  m_s = mlp_at(102), m_v = mlp_at(114), m_x = mlp_at(126);

  // ---------------- workspace layout (bump allocator, 256B aligned) --------
  char* base = (char*)d_ws;
  size_t off = 0;
  auto alloc = [&](size_t bytes) -> void* {
    off = (off + 255) & ~(size_t)255;
    void* p = base + off;
    off += bytes;
    return p;
  };
  float* t1    = (float*)alloc(NF * 4);
  float* t2    = (float*)alloc(NF * 4);
  float* xh    = (float*)alloc(NF * 4);
  float* xx    = (float*)alloc(NF * 4);
  float* xsb   = (float*)alloc(NF * 4);
  float* xpb   = (float*)alloc(NF * 4);
  float* xdb   = (float*)alloc(NF * 4);
  float* qb    = (float*)alloc(NF * 4);
  float* kbv   = (float*)alloc(NF * 4);
  float* vbv   = (float*)alloc(NF * 4);
  float* s_acc = (float*)alloc(NF * 4);
  float* ub    = (float*)alloc(NF * 4);       // later reused as y = xh+l+nl
  float* locb  = (float*)alloc(NF * 4);
  float* Qp    = (float*)alloc(NF * 4);
  float* Kp    = (float*)alloc(NF * 4);
  float* nlb   = (float*)alloc(NF * 4);
  float* pp    = (float*)alloc((size_t)NN * 3 * 128 * 4);
  float* dd    = (float*)alloc((size_t)NN * 5 * 128 * 4);
  float* pt    = (float*)alloc((size_t)NN * 3 * 256 * 4);
  float* dt    = (float*)alloc((size_t)NN * 5 * 256 * 4);
  float* hq    = (float*)alloc(NN * 4);
  float* hk    = (float*)alloc(NN * 4);
  float* rmq   = (float*)alloc(NN * 4);
  float* rmk   = (float*)alloc(NN * 4);
  float* nv    = (float*)alloc(NN * 4);
  float* smax  = (float*)alloc(BB * 4);
  float* Ksum  = (float*)alloc(BB * 128 * 4);
  int*   offs  = (int*)alloc(16 * 4);
  bf16_t* KpT  = (bf16_t*)alloc((size_t)128 * NN * 2);
  bf16_t* VT   = (bf16_t*)alloc((size_t)128 * NN * 2);
  bf16_t* KVT  = (bf16_t*)alloc((size_t)BB * 128 * 128 * 2);
  bf16_t* omgb = (bf16_t*)alloc(16384 * 2);
  bf16_t* prpb = (bf16_t*)alloc(32768 * 2);
  bf16_t* prdb = (bf16_t*)alloc(32768 * 2);
  bf16_t* rsb  = (bf16_t*)alloc(4096 * 2);
  bf16_t* rpb  = (bf16_t*)alloc(4096 * 2);
  bf16_t* rdb  = (bf16_t*)alloc(4096 * 2);
  auto allocW = [&]() -> bf16_t* { return (bf16_t*)alloc(16384 * 2); };
  MlpB w_x  = {allocW(), allocW(), allocW()};
  MlpB w_s  = {allocW(), allocW(), allocW()};
  MlpB w_p  = {allocW(), allocW(), allocW()};
  MlpB w_d  = {allocW(), allocW(), allocW()};
  MlpB w_loc= {allocW(), allocW(), allocW()};
  MlpB w_q  = {allocW(), allocW(), allocW()};
  MlpB w_k  = {allocW(), allocW(), allocW()};
  MlpB w_v  = {allocW(), allocW(), allocW()};
  MlpB w_out= {allocW(), allocW(), allocW()};
  bf16_t* preW1 = allocW();  bf16_t* preW2 = allocW();
  bf16_t* postW1 = allocW(); bf16_t* postW2 = allocW();

  float* x2_out = (float*)d_out;            // first tuple element
  float* y_out  = (float*)d_out + NF;       // second tuple element

  // ---------------- weight preparation (f32 -> bf16) -----------------------
  auto cvt = [&](const float* s, bf16_t* d, int n) {
    cvt_kernel<<<(n + 255) / 256, 256, 0, stream>>>(s, d, n);
  };
  auto cvt_mlp = [&](const MlpP& p, const MlpB& w) {
    cvt(p.W1, w.W1b, 16384); cvt(p.W2, w.W2b, 16384); cvt(p.W, w.Wb, 16384);
  };
  cvt_mlp(m_x, w_x);   cvt_mlp(m_s, w_s);   cvt_mlp(m_p, w_p);
  cvt_mlp(m_d, w_d);   cvt_mlp(m_loc, w_loc); cvt_mlp(m_q, w_q);
  cvt_mlp(m_k, w_k);   cvt_mlp(m_v, w_v);   cvt_mlp(m_out, w_out);
  cvt(b_pre.W1, preW1, 16384);  cvt(b_pre.W2, preW2, 16384);
  cvt(b_post.W1, postW1, 16384); cvt(b_post.W2, postW2, 16384);
  cvt(projp, prpb, 32768); cvt(projd, prdb, 32768);
  cvt_omega_kernel<<<64, 256, 0, stream>>>(omega, omgb, powf(128.0f, -0.25f));
  cvt_radial_kernel<<<16, 256, 0, stream>>>(rads, rsb);
  cvt_radial_kernel<<<16, 256, 0, stream>>>(radp, rpb);
  cvt_radial_kernel<<<16, 256, 0, stream>>>(radd, rdb);
  seg_offsets_kernel<<<1, 32, 0, stream>>>(bseg, offs);

  // gemm variants (templated: swish+bias [+res], or plain)
  auto gemm_act = [&](const float* X, const bf16_t* Wb, const float* bias,
                      const float* sa, const float* sb, const float* res,
                      float* Y, int M) {
    dim3 g(M / 64, 1);
    if (res) gemm_kernel<true, true, true, 128><<<g, 128, 0, stream>>>(X, Wb, bias, sa, sb, res, Y);
    else     gemm_kernel<true, true, false, 128><<<g, 128, 0, stream>>>(X, Wb, bias, sa, sb, res, Y);
  };
  auto gemm_plain128 = [&](const float* X, const bf16_t* Wb, float* Y, int M) {
    gemm_kernel<false, false, false, 128><<<dim3(M / 64, 1), 128, 0, stream>>>(
        X, Wb, nullptr, nullptr, nullptr, nullptr, Y);
  };
  auto gemm_plain256 = [&](const float* X, const bf16_t* Wb, float* Y, int M) {
    gemm_kernel<false, false, false, 256><<<dim3(M / 64, 2), 128, 0, stream>>>(
        X, Wb, nullptr, nullptr, nullptr, nullptr, Y);
  };
  auto run_mlp = [&](const MlpP& p, const MlpB& w, const float* X, float* out) {
    gemm_act(X,  w.W1b, p.c1, p.a1, p.b1, nullptr, t1, NN);
    gemm_act(t1, w.W2b, p.c2, p.a2, p.b2, X,       t2, NN);
    gemm_act(t2, w.Wb,  p.c,  p.a,  p.b,  nullptr, out, NN);
  };

  // ---------------- 1. xh = res_block(x, res_pre) --------------------------
  gemm_act(x_in, preW1, b_pre.c1, b_pre.a1, b_pre.b1, nullptr, t1, NN);
  gemm_act(t1,   preW2, b_pre.c2, b_pre.a2, b_pre.b2, x_in,    xh, NN);

  // ---------------- 2. seven res_mlps on xh --------------------------------
  run_mlp(m_x, w_x, xh, xx);
  run_mlp(m_s, w_s, xh, xsb);
  run_mlp(m_p, w_p, xh, xpb);
  run_mlp(m_d, w_d, xh, xdb);
  run_mlp(m_q, w_q, xh, qb);
  run_mlp(m_k, w_k, xh, kbv);
  run_mlp(m_v, w_v, xh, vbv);

  // ---------------- 3. pair phase ------------------------------------------
  copy_kernel<<<NF / 256, 256, 0, stream>>>(xx, s_acc, (int)NF);
  zero_kernel<<<(NN * 3 * 128) / 256, 256, 0, stream>>>(pp, NN * 3 * 128);
  zero_kernel<<<(NN * 5 * 128) / 256, 256, 0, stream>>>(dd, NN * 5 * 128);
  pair_kernel<<<PP_ / 64, 128, 0, stream>>>(rbf, idx_i, idx_j, pij, dij,
                                            xsb, xpb, xdb, rsb, rpb, rdb,
                                            s_acc, pp, dd);

  // ---------------- 4. projections + combine -------------------------------
  gemm_plain256(pp, prpb, pt, NN * 3);
  gemm_plain256(dd, prdb, dt, NN * 5);
  combine_u_kernel<<<NF / 256, 256, 0, stream>>>(s_acc, pt, dt, ub);

  // ---------------- 5. local = res_mlp(u, res_local) -----------------------
  run_mlp(m_loc, w_loc, ub, locb);

  // ---------------- 6. attention -------------------------------------------
  gemm_plain128(qb,  omgb, t1, NN);   // Uq
  gemm_plain128(kbv, omgb, t2, NN);   // Uk
  rowstats_kernel<<<NN / 4, 128, 0, stream>>>(qb, kbv, t1, t2, hq, hk, rmq, rmk);
  segmax_kernel<<<1, 256, 0, stream>>>(rmk, offs, smax);
  phiq_kernel<<<NF / 256, 256, 0, stream>>>(t1, hq, rmq, Qp);
  phik_kernel<<<NF / 256, 256, 0, stream>>>(t2, hk, smax, bseg, Kp, KpT);
  vt_kernel<<<NF / 256, 256, 0, stream>>>(vbv, VT);
  ksum_kernel<<<BB, 128, 0, stream>>>(Kp, offs, Ksum);
  norm_kernel<<<NN / 256, 256, 0, stream>>>(Qp, Ksum, bseg, nv);
  kv_kernel<<<BB, 256, 0, stream>>>(KpT, VT, offs, KVT);
  att_kernel<<<dim3(BB, 128), 128, 0, stream>>>(Qp, KVT, nv, offs, nlb);

  // ---------------- 7. x2 = res_block(xh + l + nl, res_post) ---------------
  add3_kernel<<<NF / 256, 256, 0, stream>>>(xh, locb, nlb, ub, (int)NF);
  gemm_act(ub, postW1, b_post.c1, b_post.a1, b_post.b1, nullptr, t1, NN);
  gemm_act(t1, postW2, b_post.c2, b_post.a2, b_post.b2, ub,   x2_out, NN);

  // ---------------- 8. out = res_mlp(x2, res_out) --------------------------
  gemm_act(x2_out, w_out.W1b, m_out.c1, m_out.a1, m_out.b1, nullptr, t1, NN);
  gemm_act(t1,     w_out.W2b, m_out.c2, m_out.a2, m_out.b2, x2_out,  t2, NN);
  gemm_act(t2,     w_out.Wb,  m_out.c,  m_out.a,  m_out.b,  nullptr, y_out, NN);
}
